// STGCN_66417374265406
// MI455X (gfx1250) — matrix operand order repaired and checked by reading, not compiled, not run on hardware
//
#include <hip/hip_runtime.h>
#include <hip/hip_bf16.h>
#include <math.h>

typedef __attribute__((ext_vector_type(16))) __bf16 v16bf;
typedef __attribute__((ext_vector_type(8)))  float  v8f;

#define NB      4096           // nodes
#define TP      24             // time patches
#define DM      64             // model dim
#define CM      32             // mid channels
#define BSZ     8              // batch
#define ROWS    (BSZ*CM*TP)    // 6144 GEMM rows
#define KTOT    (3*NB)         // 12288 GEMM K

static __device__ __forceinline__ unsigned short f2bf(float f) {
  unsigned u = __float_as_uint(f);
  u = (u + 0x7FFFu + ((u >> 16) & 1u)) >> 16;
  return (unsigned short)u;
}
static __device__ __forceinline__ float sigf(float x) { return 1.f / (1.f + expf(-x)); }

// ---- bf16 graph operator in B-fragment layout: Gbf[(n*3 + k)*NB + m] = G[k][n][m] ----
// Coalesced on both sides (m innermost); row length per n is KTOT, so GEMM B staging
// becomes contiguous b128 loads/stores identical to A staging.
__global__ void k_gbf(const float* __restrict__ G, unsigned short* __restrict__ Gbf) {
  const int idx = blockIdx.x * 256 + threadIdx.x;      // one thread per 4 elements
  const int m4 = (idx & (NB / 4 - 1)) * 4;
  const int n  = (idx >> 10) & (NB - 1);
  const int k  = idx >> 22;                            // 0..2
  const float4 g = *reinterpret_cast<const float4*>(&G[((size_t)k * NB + n) * NB + m4]);
  union { unsigned short s[4]; uint2 v; } o;
  o.s[0] = f2bf(g.x); o.s[1] = f2bf(g.y); o.s[2] = f2bf(g.z); o.s[3] = f2bf(g.w);
  *reinterpret_cast<uint2*>(&Gbf[((size_t)n * 3 + k) * NB + m4]) = o.v;
}

// ---------------- patch embedding + positional encoding -> h (b,64,24,4096) --------
__global__ void k_embed(const float* __restrict__ x, const float* __restrict__ W,
                        float* __restrict__ h) {
  const int idx = blockIdx.x * 256 + threadIdx.x;      // ((b*64+d)*24+t)*4096+n
  const int n = idx & (NB - 1);
  const int t = (idx >> 12) % TP;
  const int d = (idx >> 12) / TP % DM;
  const int b = idx / (NB * TP * DM);
  const float* xb = x + (size_t)b * 288 * NB + (size_t)(t * 12) * NB + n;
  float acc = 0.f;
#pragma unroll
  for (int p = 0; p < 12; p++) acc += xb[(size_t)p * NB] * W[p * DM + d];
  const float ang = (float)t * expf(-(float)(d & ~1) * (logf(10000.f) / 64.f));
  acc += (d & 1) ? cosf(ang) : sinf(ang);
  h[idx] = acc;
}

// ---------------- temporal conv (kt=3, pad 1) + GLU; align conv or identity --------
__global__ void k_tconv_glu(const float* __restrict__ in, const float* __restrict__ w,
                            const float* __restrict__ bias, const float* __restrict__ aw,
                            const float* __restrict__ ab, float* __restrict__ out,
                            int cin, int cout) {
  const int idx = blockIdx.x * 256 + threadIdx.x;      // ((b*cout+o)*24+t)*4096+n
  const int n  = idx & (NB - 1);
  const int t  = (idx >> 12) % TP;
  const int oc = (idx >> 12) / TP % cout;
  const int b  = (idx >> 12) / TP / cout;
  const float* xb = in + ((size_t)b * cin * TP + t) * NB + n;
  float a1 = bias[oc], a2 = bias[oc + cout], aa = 0.f;
  const float* w1 = w + (size_t)oc * cin * 3;
  const float* w2 = w + (size_t)(oc + cout) * cin * 3;
  for (int c = 0; c < cin; c++) {
    const float* xc = xb + (size_t)c * TP * NB;
    const float x0 = xc[0];
    const float xm = (t > 0)      ? xc[-(int)NB] : 0.f;
    const float xp = (t < TP - 1) ? xc[NB]       : 0.f;
    a1 += w1[c * 3 + 0] * xm + w1[c * 3 + 1] * x0 + w1[c * 3 + 2] * xp;
    a2 += w2[c * 3 + 0] * xm + w2[c * 3 + 1] * x0 + w2[c * 3 + 2] * xp;
    if (aw) aa += aw[oc * cin + c] * x0;
  }
  if (aw) aa += ab[oc];
  else    aa  = xb[(size_t)oc * TP * NB];              // identity align (cin==cout)
  out[idx] = (a1 + aa) * sigf(a2);
}

// ---------------- theta premix:  Z[r, k*4096+m] = bf16( sum_i theta[i,o,k]*x1 ) ----
__global__ void k_premix(const float* __restrict__ x1, const float* __restrict__ theta,
                         unsigned short* __restrict__ Z) {
  const int idx = blockIdx.x * 256 + threadIdx.x;      // r*12288 + k*4096 + m
  const int m = idx & (NB - 1);
  const int k = (idx >> 12) % 3;
  const int r = idx / (3 * NB);
  const int b = r / (CM * TP);
  const int o = (r / TP) & (CM - 1);
  const int t = r % TP;
  const float* xb = x1 + ((size_t)b * CM * TP + t) * NB + m;
  float acc = 0.f;
#pragma unroll
  for (int i = 0; i < CM; i++)
    acc += theta[(i * CM + o) * 3 + k] * xb[(size_t)i * TP * NB];
  Z[idx] = f2bf(acc);
}

// -------- big GEMM (bf16 WMMA, double-buffered LDS): xs = relu(Z@B + sb + x1) ------
__global__ void __launch_bounds__(256) k_gemm_cheb(
    const unsigned short* __restrict__ Z, const unsigned short* __restrict__ Gbf,
    const float* __restrict__ sb, float* __restrict__ x1xs) {
  __shared__ __align__(16) unsigned short As[2][128 * 32];   // A[m][k]
  __shared__ __align__(16) unsigned short Bs[2][128 * 32];   // B^T[n][k]
  const int tid  = threadIdx.x;
  const int lane = tid & 31;
  const int wave = tid >> 5;
  const int wm = wave & 3, wn = wave >> 2;                   // 4x2 waves -> 128x128 tile
  const int R0 = blockIdx.y * 128;
  const int N0 = blockIdx.x * 128;
  const int khalf = lane >> 4;
  const int l16   = lane & 15;
  const int NT    = KTOT / 32;                               // 384 K-tiles

  const v8f vzero = {};
  v8f acc[2][4];
#pragma unroll
  for (int i = 0; i < 2; i++)
#pragma unroll
    for (int j = 0; j < 4; j++) acc[i][j] = vzero;

  union Frag { uint4 q[2]; v16bf v; };

  auto stage = [&](int buf, int K0) {
#pragma unroll
    for (int c = 0; c < 2; c++) {
      const int ch  = tid + c * 256;                         // 0..511 uint4 chunks
      const int row = ch >> 2, kq = ch & 3;
      *reinterpret_cast<uint4*>(&As[buf][row * 32 + kq * 8]) =
          *reinterpret_cast<const uint4*>(&Z[(size_t)(R0 + row) * KTOT + K0 + kq * 8]);
      *reinterpret_cast<uint4*>(&Bs[buf][row * 32 + kq * 8]) =
          *reinterpret_cast<const uint4*>(&Gbf[(size_t)(N0 + row) * KTOT + K0 + kq * 8]);
    }
  };

  stage(0, 0);
  __syncthreads();

  for (int kt = 0; kt < NT; kt++) {
    const int p = kt & 1;

    Frag a[2], bf[4];
#pragma unroll
    for (int fm = 0; fm < 2; fm++) {
      const int row = wm * 32 + fm * 16 + l16;               // lanes 0-15: M; khalf: K grp
      a[fm].q[0] = *reinterpret_cast<const uint4*>(&As[p][row * 32 + khalf * 8]);
      a[fm].q[1] = *reinterpret_cast<const uint4*>(&As[p][row * 32 + 16 + khalf * 8]);
    }
#pragma unroll
    for (int fn = 0; fn < 4; fn++) {
      const int col = wn * 64 + fn * 16 + l16;               // lanes index N; khalf: K grp
      bf[fn].q[0] = *reinterpret_cast<const uint4*>(&Bs[p][col * 32 + khalf * 8]);
      bf[fn].q[1] = *reinterpret_cast<const uint4*>(&Bs[p][col * 32 + 16 + khalf * 8]);
    }
#pragma unroll
    for (int fm = 0; fm < 2; fm++)
#pragma unroll
      for (int fn = 0; fn < 4; fn++)
        acc[fm][fn] = __builtin_amdgcn_wmma_f32_16x16x32_bf16(
            false, a[fm].v, false, bf[fn].v, (short)0, acc[fm][fn], false, false);

    if (kt + 1 < NT) {
      stage(1 - p, (kt + 1) * 32);                           // overlap with WMMA pipe
      if (kt + 2 < NT) {
        __builtin_prefetch(&Z[(size_t)(R0 + (tid >> 1)) * KTOT + (kt + 2) * 32], 0, 0);
        __builtin_prefetch(&Gbf[(size_t)(N0 + (tid >> 1)) * KTOT + (kt + 2) * 32], 0, 0);
      }
    }
    __syncthreads();                                         // single barrier per tile
  }

  // epilogue: D += sb[o] + x1 ; relu ; store xs over x1 (row layout matches exactly)
#pragma unroll
  for (int fm = 0; fm < 2; fm++) {
    const int rBase = R0 + wm * 32 + fm * 16 + ((lane >> 4) << 3);
#pragma unroll
    for (int fn = 0; fn < 4; fn++) {
      const int nIdx = N0 + wn * 64 + fn * 16 + (lane & 15);
#pragma unroll
      for (int v = 0; v < 8; v++) {
        const int r = rBase + v;
        const int o = (r / TP) & (CM - 1);
        const size_t off = (size_t)r * NB + nIdx;
        const float val = acc[fm][fn][v] + sb[o] + x1xs[off];
        x1xs[off] = fmaxf(val, 0.f);
      }
    }
  }
}

// ---------------- temporal conv (kt=3) + channel-pad residual + relu (32->64) ------
__global__ void k_tconv_relu(const float* __restrict__ xs, const float* __restrict__ w,
                             const float* __restrict__ bias, float* __restrict__ out) {
  const int idx = blockIdx.x * 256 + threadIdx.x;       // ((b*64+o)*24+t)*4096+n
  const int n = idx & (NB - 1);
  const int t = (idx >> 12) % TP;
  const int o = (idx >> 12) / TP % DM;
  const int b = idx / (NB * TP * DM);
  const float* xb = xs + ((size_t)b * CM * TP + t) * NB + n;
  float a = bias[o];
  const float* wr = w + (size_t)o * CM * 3;
#pragma unroll
  for (int c = 0; c < CM; c++) {
    const float* xc = xb + (size_t)c * TP * NB;
    const float x0 = xc[0];
    const float xm = (t > 0)      ? xc[-(int)NB] : 0.f;
    const float xp = (t < TP - 1) ? xc[NB]       : 0.f;
    a += wr[c * 3] * xm + wr[c * 3 + 1] * x0 + wr[c * 3 + 2] * xp;
  }
  if (o < CM) a += xb[(size_t)o * TP * NB];             // zero-pad channel residual
  out[idx] = fmaxf(a, 0.f);
}

// ---------------- LayerNorm over (n,c) jointly per (b,t): stats then apply ---------
__global__ void k_ln_stats(const float* __restrict__ p, float* __restrict__ stats) {
  __shared__ float s1[256], s2[256];
  const int g = blockIdx.x, b = g / TP, t = g % TP;
  const float* base = p + ((size_t)b * DM * TP + t) * NB;
  float sum = 0.f, sq = 0.f;
  for (int i = threadIdx.x; i < DM * NB; i += 256) {
    const int c = i >> 12, n = i & (NB - 1);
    const float v = base[(size_t)c * TP * NB + n];
    sum += v; sq += v * v;
  }
  s1[threadIdx.x] = sum; s2[threadIdx.x] = sq;
  __syncthreads();
  for (int st = 128; st > 0; st >>= 1) {
    if (threadIdx.x < st) { s1[threadIdx.x] += s1[threadIdx.x + st]; s2[threadIdx.x] += s2[threadIdx.x + st]; }
    __syncthreads();
  }
  if (threadIdx.x == 0) {
    const float inv = 1.f / (float)(DM * NB);
    const float mean = s1[0] * inv;
    const float var  = s2[0] * inv - mean * mean;
    stats[2 * g] = mean; stats[2 * g + 1] = rsqrtf(var + 1e-5f);
  }
}
__global__ void k_ln_apply(float* __restrict__ p, const float* __restrict__ gam,
                           const float* __restrict__ bet, const float* __restrict__ stats) {
  const int idx = blockIdx.x * 256 + threadIdx.x;
  const int n = idx & (NB - 1);
  const int t = (idx >> 12) % TP;
  const int c = (idx >> 12) / TP % DM;
  const int b = idx / (NB * TP * DM);
  const int g = b * TP + t;
  const float v = (p[idx] - stats[2 * g]) * stats[2 * g + 1];
  p[idx] = v * gam[n * DM + c] + bet[n * DM + c];
}

// ---------------- 1x1 conv + residual + sigmoid (64->64) --------------------------
__global__ void k_conv1x1_sig(const float* __restrict__ in, const float* __restrict__ w,
                              const float* __restrict__ bias, float* __restrict__ out) {
  const int idx = blockIdx.x * 256 + threadIdx.x;
  const int n = idx & (NB - 1);
  const int t = (idx >> 12) % TP;
  const int o = (idx >> 12) / TP % DM;
  const int b = idx / (NB * TP * DM);
  const float* xb = in + ((size_t)b * DM * TP + t) * NB + n;
  float a = bias[o] + xb[(size_t)o * TP * NB];
#pragma unroll 8
  for (int c = 0; c < DM; c++) a += w[o * DM + c] * xb[(size_t)c * TP * NB];
  out[idx] = sigf(a);
}

// ---------------- final fully-conv over stacked (c*t): out (b,12,4096) ------------
__global__ void k_fc(const float* __restrict__ h2, const float* __restrict__ w,
                     const float* __restrict__ bias, float* __restrict__ out) {
  const int idx = blockIdx.x * 256 + threadIdx.x;       // (b*12 + j)*4096 + n
  const int n = idx & (NB - 1);
  const int j = (idx >> 12) % 12;
  const int b = idx / (NB * 12);
  const float* hb = h2 + (size_t)b * (DM * TP) * NB + n;
  const float* wr = w + j * (DM * TP);
  float a = bias[j];
#pragma unroll 8
  for (int q = 0; q < DM * TP; q++) a += wr[q] * hb[(size_t)q * NB];
  out[idx] = a;
}

extern "C" void kernel_launch(void* const* d_in, const int* in_sizes, int n_in,
                              void* d_out, int out_size, void* d_ws, size_t ws_size,
                              hipStream_t stream) {
  (void)in_sizes; (void)n_in; (void)out_size; (void)ws_size;
  const float* x  = (const float*)d_in[0];
  const float* G  = (const float*)d_in[1];
  const float* We = (const float*)d_in[3];

  const size_t S_H  = (size_t)BSZ * DM * TP * NB * 4;   // 201326592
  const size_t S_X1 = (size_t)BSZ * CM * TP * NB * 4;   // 100663296
  const size_t S_Z  = (size_t)ROWS * KTOT * 2;          // 150994944
  const size_t S_GT = (size_t)3 * NB * NB * 2;          // 100663296

  char* ws = (char*)d_ws;
  float*          Ha    = (float*)(ws);
  float*          Hb    = (float*)(ws + S_H);
  float*          X1    = (float*)(ws + 2 * S_H);
  unsigned short* Z     = (unsigned short*)(ws + 2 * S_H + S_X1);
  unsigned short* Gbf   = (unsigned short*)(ws + 2 * S_H + S_X1 + S_Z);
  float*          stats = (float*)(ws + 2 * S_H + S_X1 + S_Z + S_GT);

  const int NE64 = (BSZ * DM * TP * NB) / 256;          // 196608
  const int NE32 = (BSZ * CM * TP * NB) / 256;          // 98304
  const int NEPM = (ROWS * 3 * NB) / 256;               // 294912
  const dim3 gGemm(NB / 128, ROWS / 128);               // (32, 48)

  k_gbf<<<(3 * NB * (NB / 4)) / 256, 256, 0, stream>>>(G, Gbf);
  k_embed<<<NE64, 256, 0, stream>>>(x, We, Ha);

  // ---- ST block 1 : Ha -> Hb ----
  k_tconv_glu<<<NE32, 256, 0, stream>>>(Ha, (const float*)d_in[4], (const float*)d_in[5],
                                        (const float*)d_in[6], (const float*)d_in[7], X1, DM, CM);
  k_premix<<<NEPM, 256, 0, stream>>>(X1, (const float*)d_in[8], Z);
  k_gemm_cheb<<<gGemm, 256, 0, stream>>>(Z, Gbf, (const float*)d_in[9], X1);
  k_tconv_relu<<<NE64, 256, 0, stream>>>(X1, (const float*)d_in[10], (const float*)d_in[11], Hb);
  k_ln_stats<<<BSZ * TP, 256, 0, stream>>>(Hb, stats);
  k_ln_apply<<<NE64, 256, 0, stream>>>(Hb, (const float*)d_in[12], (const float*)d_in[13], stats);

  // ---- ST block 2 : Hb -> Ha ----
  k_tconv_glu<<<NE32, 256, 0, stream>>>(Hb, (const float*)d_in[14], (const float*)d_in[15],
                                        (const float*)d_in[16], (const float*)d_in[17], X1, DM, CM);
  k_premix<<<NEPM, 256, 0, stream>>>(X1, (const float*)d_in[18], Z);
  k_gemm_cheb<<<gGemm, 256, 0, stream>>>(Z, Gbf, (const float*)d_in[19], X1);
  k_tconv_relu<<<NE64, 256, 0, stream>>>(X1, (const float*)d_in[20], (const float*)d_in[21], Ha);
  k_ln_stats<<<BSZ * TP, 256, 0, stream>>>(Ha, stats);
  k_ln_apply<<<NE64, 256, 0, stream>>>(Ha, (const float*)d_in[22], (const float*)d_in[23], stats);

  // ---- output layer : Ha -> Hb -> Ha -> d_out ----
  k_tconv_glu<<<NE64, 256, 0, stream>>>(Ha, (const float*)d_in[24], (const float*)d_in[25],
                                        nullptr, nullptr, Hb, DM, DM);
  k_ln_stats<<<BSZ * TP, 256, 0, stream>>>(Hb, stats);
  k_ln_apply<<<NE64, 256, 0, stream>>>(Hb, (const float*)d_in[26], (const float*)d_in[27], stats);
  k_conv1x1_sig<<<NE64, 256, 0, stream>>>(Hb, (const float*)d_in[28], (const float*)d_in[29], Ha);
  k_fc<<<(BSZ * 12 * NB) / 256, 256, 0, stream>>>(Ha, (const float*)d_in[30],
                                                  (const float*)d_in[31], (float*)d_out);
}
